// RDGCNEncoder_v6_6347961663821
// MI455X (gfx1250) — matrix operand order repaired and checked by reading, not compiled, not run on hardware
//
#include <hip/hip_runtime.h>

#define N_MIR 2000
#define N_DIS 800
#define NEDGE 100000
#define EMBD  96000

typedef __bf16 bf16_t;
typedef bf16_t v16bf __attribute__((ext_vector_type(16)));
typedef float  v8f   __attribute__((ext_vector_type(8)));

typedef unsigned int uint32x4 __attribute__((ext_vector_type(4)));
typedef int          int32x8  __attribute__((ext_vector_type(8)));
typedef int          int32x4  __attribute__((ext_vector_type(4)));

struct __align__(16) U32x4 { unsigned int x, y, z, w; };
struct __align__(8)  U32x2 { unsigned int x, y; };
struct __align__(16) F32x4 { float x, y, z, w; };

// ---- Tensor Data Mover availability -------------------------------------
#if defined(__has_builtin)
#  if __has_builtin(__builtin_amdgcn_tensor_load_to_lds)
#    define USE_TDM 1
#  else
#    define USE_TDM 0
#  endif
#else
#  define USE_TDM 0
#endif
#if __has_include(<hip/amd_detail/amd_gfx1250_TDM.h>)
#  define TDM_SIX_ARGS 1
#else
#  define TDM_SIX_ARGS 0
#endif

__device__ __forceinline__ unsigned short f2bf(float f) {
    union { float f; unsigned int u; } v; v.f = f;
    unsigned int r = v.u + 0x7FFFu + ((v.u >> 16) & 1u);   // round-to-nearest-even
    return (unsigned short)(r >> 16);
}
__device__ __forceinline__ unsigned int pk2bf(float lo, float hi) {
    return (unsigned int)f2bf(lo) | ((unsigned int)f2bf(hi) << 16);
}

#if USE_TDM
// Issue one 2D TDM tile load: fp32 tensor tile -> LDS (row-major, compact).
// D# per CDNA5 ISA 8.3/8.4: group0 = {count=1, lds_addr, global_addr, type=2},
// group1 = {data_size=4B, tensor_dim0/1 (OOB zero-fill), tile_dim0/1, stride0}.
__device__ __forceinline__ void tdm_load_tile(unsigned lds_off, const float* gptr,
                                              unsigned tdim0, unsigned tdim1,
                                              unsigned tile0, unsigned tile1,
                                              unsigned long stride0_elems)
{
    unsigned long ga = (unsigned long)(size_t)gptr;
    uint32x4 g0;
    g0[0] = 1u;                                               // count=1, user mode
    g0[1] = lds_off;                                          // lds_addr (bytes)
    g0[2] = (unsigned)ga;                                     // global_addr[31:0]
    g0[3] = (unsigned)((ga >> 32) & 0x01FFFFFFu) | (2u << 30);// [56:32] | type=2
    int32x8 g1;
    g1[0] = (int)(2u << 16);                                  // data_size=2 (4 bytes)
    g1[1] = (int)((tdim0 & 0xFFFFu) << 16);                   // tensor_dim0[15:0]
    g1[2] = (int)((tdim0 >> 16) | ((tdim1 & 0xFFFFu) << 16)); // td0[31:16] | td1[15:0]
    g1[3] = (int)((tdim1 >> 16) | (tile0 << 16));             // td1[31:16] | tile_dim0
    g1[4] = (int)(tile1 & 0xFFFFu);                           // tile_dim1 (tile_dim2=0)
    g1[5] = (int)(unsigned)(stride0_elems & 0xFFFFFFFFu);     // tensor_dim0_stride[31:0]
    g1[6] = (int)(unsigned)((stride0_elems >> 32) & 0xFFFFu); // stride[47:32] (stride1=0)
    g1[7] = 0;
    int32x4 z4 = {0, 0, 0, 0};
#if TDM_SIX_ARGS
    int32x8 z8 = {0, 0, 0, 0, 0, 0, 0, 0};
    __builtin_amdgcn_tensor_load_to_lds(g0, g1, z4, z4, z8, 0);
#else
    __builtin_amdgcn_tensor_load_to_lds(g0, g1, z4, z4, 0);
#endif
}
#endif

// ---------------------------------------------------------------------------
// Tiled bf16 WMMA GEMM, split-K, atomic fp32 epilogue.
//   C[M,N] += scale * (A[M,K] @ B[K,N])   (fp32 memory -> bf16 tiles -> WMMA)
// Block: 256 threads (8 waves). Block tile 128x128, wave tile 32x64, BK=32.
// TDM path: double-buffered fp32 staging tiles DMA'd by the Tensor Data Mover
// (hardware OOB zero-fill => guard-free conversion), overlapped with WMMA.
// ---------------------------------------------------------------------------
#define BM 128
#define BN 128
#define BK 32

__global__ __launch_bounds__(256)
void gemm_bf16_wmma_atomic(const float* __restrict__ A, const float* __restrict__ B,
                           float* __restrict__ C, int M, int N, int K,
                           const float* __restrict__ scales, int sidx)
{
#if USE_TDM
    __shared__ float stgA[2][BM * BK];   // 2 x 16 KB fp32 staging (TDM dest)
    __shared__ float stgB[2][BK * BN];   // 2 x 16 KB fp32 staging (TDM dest)
#endif
    __shared__ unsigned short As[BM * BK];   // bf16 [m][k]
    __shared__ unsigned short Bs[BN * BK];   // bf16 [n][k] (transposed)

    const int tid    = threadIdx.x;
    const int lane   = tid & 31;
    const int wave   = tid >> 5;
    const int laneHi = lane >> 4;
    const int lane15 = lane & 15;

    const int mBase = blockIdx.y * BM;
    const int nBase = blockIdx.x * BN;

    const long splits = (long)gridDim.z;
    const long kPer   = (((long)K + splits * BK - 1) / (splits * BK)) * BK;
    const long k0     = (long)blockIdx.z * kPer;
    long kEnd         = k0 + kPer;
    if (kEnd > K) kEnd = K;
    if (k0 >= kEnd) return;

    const int wM = (wave >> 1) * 32;   // 0,32,64,96
    const int wN = (wave & 1) * 64;    // 0,64

    v8f acc[2][4] = {};

#if USE_TDM
    const unsigned offA[2] = { (unsigned)(size_t)&stgA[0][0], (unsigned)(size_t)&stgA[1][0] };
    const unsigned offB[2] = { (unsigned)(size_t)&stgB[0][0], (unsigned)(size_t)&stgB[1][0] };

    // prologue: DMA first tile pair
    if (wave == 0) {
        tdm_load_tile(offA[0], A + (long)mBase * K + k0,
                      (unsigned)(kEnd - k0), (unsigned)(M - mBase), BK, BM, (unsigned long)K);
        tdm_load_tile(offB[0], B + k0 * (long)N + nBase,
                      (unsigned)(N - nBase), (unsigned)(kEnd - k0), BN, BK, (unsigned long)N);
        __builtin_amdgcn_s_wait_tensorcnt(0);
    }
    __syncthreads();

    int buf = 0;
    for (long kt = k0; kt < kEnd; kt += BK) {
        // overlap: DMA next tile pair into the other staging buffer
        long kn = kt + BK;
        if (wave == 0 && kn < kEnd) {
            tdm_load_tile(offA[buf ^ 1], A + (long)mBase * K + kn,
                          (unsigned)(kEnd - kn), (unsigned)(M - mBase), BK, BM, (unsigned long)K);
            tdm_load_tile(offB[buf ^ 1], B + kn * (long)N + nBase,
                          (unsigned)(N - nBase), (unsigned)(kEnd - kn), BN, BK, (unsigned long)N);
        }

        // guard-free fp32 -> bf16 conversion out of TDM staging
        const float* sA = stgA[buf];
        const float* sB = stgB[buf];
        #pragma unroll
        for (int i = 0; i < 4; ++i) {
            int idx = tid + i * 256;
            int r   = idx >> 3;            // 0..127
            int c   = (idx & 7) << 2;      // 0..28
            F32x4 a4 = *(const F32x4*)(sA + r * BK + c);
            U32x2 p; p.x = pk2bf(a4.x, a4.y); p.y = pk2bf(a4.z, a4.w);
            *(U32x2*)(&As[r * BK + c]) = p;
        }
        #pragma unroll
        for (int i = 0; i < 4; ++i) {
            int idx = tid + i * 256;
            int kk  = idx >> 5;            // 0..31
            int n4  = (idx & 31) << 2;     // 0..124
            F32x4 b4 = *(const F32x4*)(sB + kk * BN + n4);
            Bs[(n4 + 0) * BK + kk] = f2bf(b4.x);
            Bs[(n4 + 1) * BK + kk] = f2bf(b4.y);
            Bs[(n4 + 2) * BK + kk] = f2bf(b4.z);
            Bs[(n4 + 3) * BK + kk] = f2bf(b4.w);
        }
        __syncthreads();   // bf16 tiles ready
#else
    for (long kt = k0; kt < kEnd; kt += BK) {
        __syncthreads();
        #pragma unroll
        for (int i = 0; i < 4; ++i) {
            int idx = tid + i * 256;
            int r   = idx >> 3;
            int c   = (idx & 7) << 2;
            int gm  = mBase + r;
            F32x4 a4 = {0.f, 0.f, 0.f, 0.f};
            if (gm < M && (kt + c) < kEnd)
                a4 = *(const F32x4*)(A + (long)gm * K + kt + c);
            As[r * BK + c + 0] = f2bf(a4.x);
            As[r * BK + c + 1] = f2bf(a4.y);
            As[r * BK + c + 2] = f2bf(a4.z);
            As[r * BK + c + 3] = f2bf(a4.w);
        }
        __builtin_prefetch(A + (long)(mBase + (tid >> 3)) * K + kt + BK, 0, 3);
        #pragma unroll
        for (int i = 0; i < 4; ++i) {
            int idx = tid + i * 256;
            int kk  = idx >> 5;
            int n4  = (idx & 31) << 2;
            F32x4 b4 = {0.f, 0.f, 0.f, 0.f};
            if ((kt + kk) < kEnd)
                b4 = *(const F32x4*)(B + (kt + kk) * (long)N + nBase + n4);
            Bs[(n4 + 0) * BK + kk] = f2bf(b4.x);
            Bs[(n4 + 1) * BK + kk] = f2bf(b4.y);
            Bs[(n4 + 2) * BK + kk] = f2bf(b4.z);
            Bs[(n4 + 3) * BK + kk] = f2bf(b4.w);
        }
        __syncthreads();
#endif

        // ---- fragments per the CDNA5 16-bit WMMA VGPR layouts ----
        union Frag { v16bf v; U32x4 q[2]; };
        Frag fa[2], fb[4];
        #pragma unroll
        for (int i = 0; i < 2; ++i) {
            // A: M = lane15; laneLo: K0..7 & K16..23, laneHi: K8..15 & K24..31
            int base = (wM + i * 16 + lane15) * BK + laneHi * 8;
            fa[i].q[0] = *(const U32x4*)(&As[base]);
            fa[i].q[1] = *(const U32x4*)(&As[base + 16]);
        }
        #pragma unroll
        for (int j = 0; j < 4; ++j) {
            // B: N = lane15; laneLo: K0..7 & K16..23, laneHi: K8..15 & K24..31
            int base = (wN + j * 16 + lane15) * BK + laneHi * 8;
            fb[j].q[0] = *(const U32x4*)(&Bs[base]);
            fb[j].q[1] = *(const U32x4*)(&Bs[base + 16]);
        }
        #pragma unroll
        for (int i = 0; i < 2; ++i)
            #pragma unroll
            for (int j = 0; j < 4; ++j)
                acc[i][j] = __builtin_amdgcn_wmma_f32_16x16x32_bf16(
                    false, fa[i].v, false, fb[j].v, (short)0, acc[i][j], false, false);

#if USE_TDM
        if (wave == 0) __builtin_amdgcn_s_wait_tensorcnt(0);
        __syncthreads();   // frag reads done + next staging buffer complete
        buf ^= 1;
#endif
    }

    const float s = scales ? scales[sidx] : 1.0f;
    #pragma unroll
    for (int i = 0; i < 2; ++i)
        #pragma unroll
        for (int j = 0; j < 4; ++j)
            #pragma unroll
            for (int r = 0; r < 8; ++r) {
                int row = mBase + wM + i * 16 + laneHi * 8 + r;  // C: VGPR r -> M=r / r+8
                int col = nBase + wN + j * 16 + lane15;          //    N = lane15
                if (row < M && col < N)
                    atomicAdd(&C[(long)row * N + col], s * acc[i][j][r]);
            }
}

// ---------------------------------------------------------------------------
// Small helper kernels
// ---------------------------------------------------------------------------
__global__ void softmax_weights_k(const float* mw, const float* dw, float* scales) {
    if (threadIdx.x == 0 && blockIdx.x == 0) {
        float m = fmaxf(mw[0], fmaxf(mw[1], mw[2]));
        float e0 = __expf(mw[0] - m), e1 = __expf(mw[1] - m), e2 = __expf(mw[2] - m);
        float inv = 1.f / (e0 + e1 + e2);
        scales[0] = e0 * inv; scales[1] = e1 * inv; scales[2] = e2 * inv;
        float md = fmaxf(dw[0], dw[1]);
        float f0 = __expf(dw[0] - md), f1 = __expf(dw[1] - md);
        float invd = 1.f / (f0 + f1);
        scales[3] = f0 * invd; scales[4] = f1 * invd;
    }
}

__global__ void bias_rows_k(float* C, long M, long N, const float* bias,
                            const float* scales, int sidx, int init) {
    long i = (long)blockIdx.x * blockDim.x + threadIdx.x;
    if (i >= M * N) return;
    float s = scales ? scales[sidx] : 1.f;
    float v = bias ? s * bias[i % N] : 0.f;
    C[i] = (init ? 0.f : C[i]) + v;
}

__global__ void edge_count_k(const int* __restrict__ dst, float* cnt, int E) {
    int e = blockIdx.x * blockDim.x + threadIdx.x;
    if (e < E) atomicAdd(&cnt[dst[e]], 1.0f);
}

__global__ void edge_scatter_k(const float* __restrict__ x, const int* __restrict__ src,
                               const int* __restrict__ dst, float* sums, int E, int d) {
    long i = (long)blockIdx.x * blockDim.x + threadIdx.x;
    int dq = d >> 2;
    if (i >= (long)E * dq) return;
    int e = (int)(i / dq);
    int c = (int)(i % dq) << 2;
    F32x4 v = *(const F32x4*)(x + (long)src[e] * d + c);
    float* o = sums + (long)dst[e] * d + c;
    atomicAdd(o + 0, v.x); atomicAdd(o + 1, v.y);
    atomicAdd(o + 2, v.z); atomicAdd(o + 3, v.w);
}

__global__ void mean_div_k(float* sums, const float* __restrict__ cnt, long M, int d) {
    long i = (long)blockIdx.x * blockDim.x + threadIdx.x;
    if (i >= M * d) return;
    float c = cnt[i / d];
    sums[i] = sums[i] / (c < 1.f ? 1.f : c);
}

__global__ void lrelu_k(float* x, long n) {
    long i = (long)blockIdx.x * blockDim.x + threadIdx.x;
    if (i >= n) return;
    float v = x[i];
    x[i] = (v >= 0.f) ? v : 0.01f * v;
}

// ---------------------------------------------------------------------------
extern "C" void kernel_launch(void* const* d_in, const int* in_sizes, int n_in,
                              void* d_out, int out_size, void* d_ws, size_t ws_size,
                              hipStream_t stream) {
    const float* mir_emb = (const float*)d_in[0];
    const float* mir_sim = (const float*)d_in[1];
    const float* mir_ass = (const float*)d_in[2];
    const float* dis_sim = (const float*)d_in[3];
    const float* dis_ass = (const float*)d_in[4];
    const int*   src_m2d = (const int*)d_in[5];
    const int*   dst_m2d = (const int*)d_in[6];
    const int*   src_d2m = (const int*)d_in[7];
    const int*   dst_d2m = (const int*)d_in[8];
    const float* W_emb  = (const float*)d_in[9];
    const float* b_emb  = (const float*)d_in[10];
    const float* W_msim = (const float*)d_in[11];
    const float* b_msim = (const float*)d_in[12];
    const float* W_mass = (const float*)d_in[13];
    const float* b_mass = (const float*)d_in[14];
    const float* W_dsim = (const float*)d_in[15];
    const float* b_dsim = (const float*)d_in[16];
    const float* W_dass = (const float*)d_in[17];
    const float* b_dass = (const float*)d_in[18];
    const float* mir_w  = (const float*)d_in[19];
    const float* dis_w  = (const float*)d_in[20];

    struct ET { const float *Wl, *bl, *Wr; };
    ET P[3][2];   // [layer][0=m2d, 1=d2m]
    int ii = 21;
    for (int l = 0; l < 3; ++l)
        for (int et = 0; et < 2; ++et) {
            P[l][et].Wl = (const float*)d_in[ii++];
            P[l][et].bl = (const float*)d_in[ii++];
            P[l][et].Wr = (const float*)d_in[ii++];
        }

    float* ws = (float*)d_ws;
    size_t off = 0;
    auto alloc = [&](size_t n) { float* p = ws + off; off += (n + 255) & ~(size_t)255; return p; };
    float* scales = alloc(8);
    float* xm[2]  = { alloc((size_t)N_MIR * 512), alloc((size_t)N_MIR * 512) };
    float* xd[2]  = { alloc((size_t)N_DIS * 512), alloc((size_t)N_DIS * 512) };
    float* sum_m  = alloc((size_t)N_MIR * 512);
    float* sum_d  = alloc((size_t)N_DIS * 512);
    float* cnt_m  = alloc(N_MIR);
    float* cnt_d  = alloc(N_DIS);

    auto gemm = [&](const float* A, const float* B, float* C, int M, int N, int K,
                    const float* sc, int sidx, int splits) {
        dim3 grid(N / 128, (M + 127) / 128, splits);
        gemm_bf16_wmma_atomic<<<grid, 256, 0, stream>>>(A, B, C, M, N, K, sc, sidx);
    };
    auto bias = [&](float* C, long M, long N, const float* b, const float* sc, int sidx, int init) {
        long tot = M * N;
        bias_rows_k<<<(int)((tot + 255) / 256), 256, 0, stream>>>(C, M, N, b, sc, sidx, init);
    };

    // ---- softmax attention weights ----
    softmax_weights_k<<<1, 32, 0, stream>>>(mir_w, dis_w, scales);

    // ---- input projections ----
    float *cm = xm[0], *cd = xd[0];
    bias(cm, N_MIR, 256, b_msim, scales, 0, 1);
    bias(cm, N_MIR, 256, b_mass, scales, 1, 0);
    bias(cm, N_MIR, 256, b_emb,  scales, 2, 0);
    gemm(mir_sim, W_msim, cm, N_MIR, 256, N_MIR, scales, 0, 4);
    gemm(mir_ass, W_mass, cm, N_MIR, 256, N_DIS, scales, 1, 2);
    gemm(mir_emb, W_emb,  cm, N_MIR, 256, EMBD,  scales, 2, 32);   // the big one: split-K 32

    bias(cd, N_DIS, 256, b_dsim, scales, 3, 1);
    bias(cd, N_DIS, 256, b_dass, scales, 4, 0);
    gemm(dis_sim, W_dsim, cd, N_DIS, 256, N_DIS, scales, 3, 2);
    gemm(dis_ass, W_dass, cd, N_DIS, 256, N_MIR, scales, 4, 4);

    // ---- 3 hetero SAGE layers ----
    const int din[3]  = {256, 512, 256};
    const int dout[3] = {512, 256, 128};
    float* out_mir = (float*)d_out;
    float* out_dis = (float*)d_out + (size_t)N_MIR * 128;

    for (int l = 0; l < 3; ++l) {
        int di = din[l], dd = dout[l];
        float* nm = (l == 2) ? out_mir : xm[(l + 1) & 1];
        float* nd = (l == 2) ? out_dis : xd[(l + 1) & 1];

        // ---- m2d: mean over mir neighbors -> new_dis ----
        bias(sum_d, N_DIS, di, nullptr, nullptr, 0, 1);
        bias(cnt_d, 1, N_DIS, nullptr, nullptr, 0, 1);
        edge_count_k<<<(NEDGE + 255) / 256, 256, 0, stream>>>(dst_m2d, cnt_d, NEDGE);
        {
            long tot = (long)NEDGE * (di >> 2);
            edge_scatter_k<<<(int)((tot + 255) / 256), 256, 0, stream>>>(cm, src_m2d, dst_m2d, sum_d, NEDGE, di);
            long tm = (long)N_DIS * di;
            mean_div_k<<<(int)((tm + 255) / 256), 256, 0, stream>>>(sum_d, cnt_d, N_DIS, di);
        }
        bias(nd, N_DIS, dd, P[l][0].bl, nullptr, 0, 1);
        gemm(sum_d, P[l][0].Wl, nd, N_DIS, dd, di, nullptr, 0, 2);
        gemm(cd,    P[l][0].Wr, nd, N_DIS, dd, di, nullptr, 0, 2);

        // ---- d2m: mean over dis neighbors -> new_mir ----
        bias(sum_m, N_MIR, di, nullptr, nullptr, 0, 1);
        bias(cnt_m, 1, N_MIR, nullptr, nullptr, 0, 1);
        edge_count_k<<<(NEDGE + 255) / 256, 256, 0, stream>>>(dst_d2m, cnt_m, NEDGE);
        {
            long tot = (long)NEDGE * (di >> 2);
            edge_scatter_k<<<(int)((tot + 255) / 256), 256, 0, stream>>>(cd, src_d2m, dst_d2m, sum_m, NEDGE, di);
            long tm = (long)N_MIR * di;
            mean_div_k<<<(int)((tm + 255) / 256), 256, 0, stream>>>(sum_m, cnt_m, N_MIR, di);
        }
        bias(nm, N_MIR, dd, P[l][1].bl, nullptr, 0, 1);
        gemm(sum_m, P[l][1].Wl, nm, N_MIR, dd, di, nullptr, 0, 2);
        gemm(cm,    P[l][1].Wr, nm, N_MIR, dd, di, nullptr, 0, 2);

        // ---- leaky relu ----
        {
            long n1 = (long)N_MIR * dd, n2 = (long)N_DIS * dd;
            lrelu_k<<<(int)((n1 + 255) / 256), 256, 0, stream>>>(nm, n1);
            lrelu_k<<<(int)((n2 + 255) / 256), 256, 0, stream>>>(nd, n2);
        }
        cm = nm; cd = nd;
    }
}
